// MultiHeadSelfAttention_25056839205789
// MI455X (gfx1250) — matrix operand order repaired and checked by reading, not compiled, not run on hardware
//
#include <hip/hip_runtime.h>
#include <hip/hip_bf16.h>

// ---- CDNA5 / gfx1250 wave32 WMMA attention ----
#define TT 4096
#define DD 2048
#define HH 16
#define HD 128
#define D3 (3 * DD)

typedef __attribute__((ext_vector_type(16))) __bf16 v16bf;
typedef __attribute__((ext_vector_type(8)))  __bf16 v8bf;
typedef __attribute__((ext_vector_type(8)))  float  v8f;

static __device__ inline v8f v8f_zero() {
  v8f z;
#pragma unroll
  for (int i = 0; i < 8; ++i) z[i] = 0.0f;
  return z;
}

static __device__ inline v16bf mk16(v8bf lo, v8bf hi) {
  v16bf r;
#pragma unroll
  for (int i = 0; i < 8; ++i) { r[i] = lo[i]; r[i + 8] = hi[i]; }
  return r;
}

static __device__ inline v16bf load_frag(const __bf16* p0, const __bf16* p1) {
  return mk16(*(const v8bf*)p0, *(const v8bf*)p1);
}

// Async 32-byte global->LDS copy (gfx1250 VGLOBAL async path, ASYNCcnt).
// IOFFSET is added to BOTH the LDS and the global address, so one address
// pair covers both 16B transfers.
static __device__ inline void async_copy32(unsigned lds_off,
                                           const __bf16* gaddr) {
  asm volatile(
      "global_load_async_to_lds_b128 %0, %1, off\n\t"
      "global_load_async_to_lds_b128 %0, %1, off offset:16"
      :
      : "v"(lds_off), "v"(gaddr)
      : "memory");
}

static __device__ inline void wait_async0() {
  asm volatile("s_wait_asynccnt 0x0" ::: "memory");
}

static __device__ inline void wave_lds_fence() {
  asm volatile("s_wait_dscnt 0x0" ::: "memory");
}

// ---------------------------------------------------------------------------
// Kernel 1: f32 -> bf16 convert
// ---------------------------------------------------------------------------
__global__ void f32_to_bf16_kernel(const float* __restrict__ in,
                                   __bf16* __restrict__ out, int n) {
  int i = blockIdx.x * blockDim.x + threadIdx.x;
  if (i < n) out[i] = (__bf16)in[i];
}

// ---------------------------------------------------------------------------
// Kernel 2: bf16 GEMM, f32 accumulate/output.  A: MxK row-major, B: KxN
// row-major, C: MxN f32.  Block tile 128x128, 8 waves; each wave does a
// 64x32 region = 4x2 WMMA 16x16 tiles.  A/B K-slices are staged in LDS with
// DOUBLE-BUFFERED async global->LDS copies: tile k+1 streams in (ASYNCcnt)
// while WMMAs consume tile k; one barrier per K-step.  Safe because a wave
// only signals the barrier after its WMMAs forced s_wait_dscnt on the
// ds_load fragments of the current buffer.
// ---------------------------------------------------------------------------
__global__ __launch_bounds__(256) void gemm_bf16_wmma(
    const __bf16* __restrict__ A, const __bf16* __restrict__ B,
    float* __restrict__ C, int M, int N, int K) {
  __shared__ __bf16 ldsA[2][128][32];  // [buf][m][k]  2 x 8KB
  __shared__ __bf16 ldsB[2][32][128];  // [buf][k][n]  2 x 8KB

  const int tid = threadIdx.x;
  const int wave = tid >> 5;
  const int lane = tid & 31;
  const int l15 = lane & 15;
  const int kg = lane >> 4;  // K-group half of the wave

  const int mblk = blockIdx.y * 128;
  const int nblk = blockIdx.x * 128;
  const int m0w = (wave >> 2) * 64;  // wave's 4 M-units within block
  const int n0w = (wave & 3) * 32;   // wave's 2 N-units within block

  // cooperative tile-copy assignment: 32 bytes per thread per tile
  const int arow = tid >> 1, ak = (tid & 1) * 16;   // A: 128 rows x 2 chunks
  const int brow = tid >> 3, bn = (tid & 7) * 16;   // B: 32 rows x 8 chunks
  unsigned laOff[2], lbOff[2];
#pragma unroll
  for (int b = 0; b < 2; ++b) {
    laOff[b] = (unsigned)(size_t)&ldsA[b][arow][ak];
    lbOff[b] = (unsigned)(size_t)&ldsB[b][brow][bn];
  }
  const __bf16* gA = A + (size_t)(mblk + arow) * K + ak;
  const __bf16* gB = B + (size_t)brow * N + nblk + bn;

  v8f acc[4][2];
#pragma unroll
  for (int mu = 0; mu < 4; ++mu)
#pragma unroll
    for (int nu = 0; nu < 2; ++nu) acc[mu][nu] = v8f_zero();

  // prologue: stage first K-slice into buffer 0
  async_copy32(laOff[0], gA);
  async_copy32(lbOff[0], gB);

  for (int k0 = 0; k0 < K; k0 += 32) {
    const int cur = (k0 >> 5) & 1;

    wait_async0();     // this wave's copies into buf[cur] landed
    __syncthreads();   // everyone's copies landed; buf[cur^1] free to refill

    if (k0 + 32 < K) {  // copy-ahead: stream tile k+1 while computing tile k
      async_copy32(laOff[cur ^ 1], gA + (k0 + 32));
      async_copy32(lbOff[cur ^ 1], gB + (size_t)(k0 + 32) * N);
    }

    // ---- fragments from LDS ----
    v16bf aF[4];
#pragma unroll
    for (int mu = 0; mu < 4; ++mu) {
      const __bf16* ap = &ldsA[cur][m0w + mu * 16 + l15][8 * kg];
      aF[mu] = load_frag(ap, ap + 16);  // K = {8kg..+7, 16+8kg..+7}
    }
    v16bf bF[2];
#pragma unroll
    for (int nu = 0; nu < 2; ++nu) {
      const __bf16* bp = &ldsB[cur][8 * kg][n0w + nu * 16 + l15];
#pragma unroll
      for (int s = 0; s < 16; ++s)  // N = l15, K = {8kg..+7, 16+8kg..+7}
        bF[nu][s] = bp[(s < 8 ? s : 8 + s) * 128];
    }
#pragma unroll
    for (int mu = 0; mu < 4; ++mu)
#pragma unroll
      for (int nu = 0; nu < 2; ++nu)
        acc[mu][nu] = __builtin_amdgcn_wmma_f32_16x16x32_bf16(
            false, aF[mu], false, bF[nu], (short)0, acc[mu][nu], false, false);
  }

  // ---- epilogue ----
#pragma unroll
  for (int mu = 0; mu < 4; ++mu)
#pragma unroll
    for (int nu = 0; nu < 2; ++nu) {
      float* cp = C + (size_t)(mblk + m0w + mu * 16 + 8 * kg) * N + nblk +
                  n0w + nu * 16 + l15;
#pragma unroll
      for (int r = 0; r < 8; ++r) {
        *cp = acc[mu][nu][r];
        cp += N;
      }
    }
}

// ---------------------------------------------------------------------------
// Kernel 3: per-(token, head) l2-normalize q and k, emit head-major bf16:
//   q  : [H][T][HD]       (rows for A fragments)
//   kT : [H][HD][T]       (transposed so QK^T B fragments are contiguous)
//   v  : [H][T][HD]
// One wave per (t, h); lane handles 4 of the 128 dims.
// ---------------------------------------------------------------------------
__global__ __launch_bounds__(256) void qk_norm_kernel(
    const float* __restrict__ qkv, __bf16* __restrict__ q,
    __bf16* __restrict__ kT, __bf16* __restrict__ v) {
  const int wave = threadIdx.x >> 5;
  const int lane = threadIdx.x & 31;
  const int wid = blockIdx.x * 8 + wave;
  const int h = wid & (HH - 1);
  const int t = wid >> 4;
  const int d0 = lane * 4;

  const float* row = qkv + (size_t)t * D3 + h * HD + d0;
  float qv[4], kv[4], vv[4];
#pragma unroll
  for (int i = 0; i < 4; ++i) {
    qv[i] = row[i];
    kv[i] = row[DD + i];
    vv[i] = row[2 * DD + i];
  }
  float sq = 0.f, sk = 0.f;
#pragma unroll
  for (int i = 0; i < 4; ++i) { sq += qv[i] * qv[i]; sk += kv[i] * kv[i]; }
#pragma unroll
  for (int off = 16; off >= 1; off >>= 1) {
    sq += __shfl_xor(sq, off, 32);
    sk += __shfl_xor(sk, off, 32);
  }
  const float rq = 1.0f / fmaxf(sqrtf(sq), 1e-12f);
  const float rk = 1.0f / fmaxf(sqrtf(sk), 1e-12f);

  __bf16* qp = q + ((size_t)h * TT + t) * HD + d0;
  __bf16* vp = v + ((size_t)h * TT + t) * HD + d0;
#pragma unroll
  for (int i = 0; i < 4; ++i) {
    qp[i] = (__bf16)(qv[i] * rq);
    vp[i] = (__bf16)vv[i];
    kT[((size_t)h * HD + d0 + i) * TT + t] = (__bf16)(kv[i] * rk);
  }
}

// ---------------------------------------------------------------------------
// Kernel 4: flash attention.  One wave owns 16 query rows of one head and
// streams all keys in chunks of 32 with online softmax (f32 stats).
// QK^T: 8 WMMAs / chunk; P re-layout (C-frag -> A-frag) via a PRIVATE
// per-wave LDS slice: LDS is in-order per wave, so a wave-local dscnt fence
// suffices -- no block-wide barrier, waves stream independently.
// P*V: 8 WMMAs / chunk.
// ---------------------------------------------------------------------------
__global__ __launch_bounds__(256) void flash_attn_wmma(
    const __bf16* __restrict__ q, const __bf16* __restrict__ kT,
    const __bf16* __restrict__ v, const float* __restrict__ mask,
    __bf16* __restrict__ out) {
  __shared__ __bf16 ldsP[8][16][32];  // per-wave 16x32 P tile (1 KB each)

  const int wave = threadIdx.x >> 5;
  const int lane = threadIdx.x & 31;
  const int l15 = lane & 15;
  const int kg = lane >> 4;
  const int wid = blockIdx.x * 8 + wave;
  const int h = wid >> 8;            // wid / (T/16)
  const int t0 = (wid & 255) * 16;   // query tile base
  const float scale = 0.088388347648318447f;  // 1/sqrt(HD)

  // Q fragments: 4 chunks of K=32 covering HD=128
  v16bf aQ[4];
#pragma unroll
  for (int dc = 0; dc < 4; ++dc) {
    const __bf16* qp =
        q + ((size_t)h * TT + t0 + l15) * HD + dc * 32 + 8 * kg;
    aQ[dc] = load_frag(qp, qp + 16);
  }

  float mstat[8], lstat[8];
  v8f oacc[8];
#pragma unroll
  for (int r = 0; r < 8; ++r) { mstat[r] = -1e30f; lstat[r] = 0.f; }
#pragma unroll
  for (int dc = 0; dc < 8; ++dc) oacc[dc] = v8f_zero();

  for (int s0 = 0; s0 < TT; s0 += 32) {
    // ---- scores: two 16x16 tiles over s0..s0+31 ----
    v8f c0 = v8f_zero(), c1 = v8f_zero();
#pragma unroll
    for (int dc = 0; dc < 4; ++dc) {
      const __bf16* kp =
          kT + ((size_t)h * HD + dc * 32 + 8 * kg) * TT + s0 + l15;
      v16bf b0, b1;
#pragma unroll
      for (int s = 0; s < 16; ++s) {
        b0[s] = kp[(size_t)(s < 8 ? s : 8 + s) * TT];
        b1[s] = kp[(size_t)(s < 8 ? s : 8 + s) * TT + 16];
      }
      c0 = __builtin_amdgcn_wmma_f32_16x16x32_bf16(false, aQ[dc], false, b0,
                                                   (short)0, c0, false, false);
      c1 = __builtin_amdgcn_wmma_f32_16x16x32_bf16(false, aQ[dc], false, b1,
                                                   (short)0, c1, false, false);
    }
    // ---- scale + mask + online softmax (per row r: M = r + 8*kg) ----
#pragma unroll
    for (int r = 0; r < 8; ++r) {
      const float* mrow = mask + (size_t)(t0 + r + 8 * kg) * TT + s0;
      c0[r] = c0[r] * scale + mrow[l15];
      c1[r] = c1[r] * scale + mrow[16 + l15];
      float vmax = fmaxf(c0[r], c1[r]);
#pragma unroll
      for (int off = 8; off >= 1; off >>= 1)
        vmax = fmaxf(vmax, __shfl_xor(vmax, off, 32));  // stays in 16-lane half
      const float mn = fmaxf(mstat[r], vmax);
      const float resc = __expf(mstat[r] - mn);
      mstat[r] = mn;
      const float p0 = __expf(c0[r] - mn);
      const float p1 = __expf(c1[r] - mn);
      c0[r] = p0;
      c1[r] = p1;
      float rs = p0 + p1;
#pragma unroll
      for (int off = 8; off >= 1; off >>= 1) rs += __shfl_xor(rs, off, 32);
      lstat[r] = lstat[r] * resc + rs;
#pragma unroll
      for (int dc = 0; dc < 8; ++dc) oacc[dc][r] *= resc;
    }
    // ---- re-layout P (C-fragment -> A-fragment) via private LDS slice ----
#pragma unroll
    for (int r = 0; r < 8; ++r) {
      const int m = r + 8 * kg;
      ldsP[wave][m][l15] = (__bf16)c0[r];
      ldsP[wave][m][16 + l15] = (__bf16)c1[r];
    }
    wave_lds_fence();  // wave-local: LDS in-order per wave, no block barrier
    const __bf16* pb = &ldsP[wave][l15][0];
    v16bf aP = load_frag(pb + 8 * kg, pb + 16 + 8 * kg);
    // ---- O += P * V : 8 N-tiles of 16 covering HD=128 ----
#pragma unroll
    for (int dc = 0; dc < 8; ++dc) {
      const __bf16* vp =
          v + ((size_t)h * TT + s0 + 8 * kg) * HD + dc * 16 + l15;
      v16bf bV;
#pragma unroll
      for (int s = 0; s < 16; ++s) bV[s] = vp[(s < 8 ? s : 8 + s) * HD];
      oacc[dc] = __builtin_amdgcn_wmma_f32_16x16x32_bf16(
          false, aP, false, bV, (short)0, oacc[dc], false, false);
    }
  }

  // ---- epilogue: divide by softmax denom, write token-major bf16 ----
#pragma unroll
  for (int dc = 0; dc < 8; ++dc)
#pragma unroll
    for (int r = 0; r < 8; ++r) {
      const int t = t0 + r + 8 * kg;
      out[(size_t)t * DD + h * HD + dc * 16 + l15] =
          (__bf16)(oacc[dc][r] / lstat[r]);
    }
}

// ---------------------------------------------------------------------------
extern "C" void kernel_launch(void* const* d_in, const int* in_sizes, int n_in,
                              void* d_out, int out_size, void* d_ws,
                              size_t ws_size, hipStream_t stream) {
  const float* x = (const float*)d_in[0];
  const float* attn_mask = (const float*)d_in[1];
  const float* W_qkv = (const float*)d_in[2];
  const float* W_out = (const float*)d_in[3];
  float* outp = (float*)d_out;

  // workspace partition (all 256B-aligned sizes)
  char* w = (char*)d_ws;
  __bf16* xb = (__bf16*)w;                 w += (size_t)TT * DD * 2;
  __bf16* wqkvb = (__bf16*)w;              w += (size_t)DD * D3 * 2;
  __bf16* woutb = (__bf16*)w;              w += (size_t)DD * DD * 2;
  float* qkv = (float*)w;                  w += (size_t)TT * D3 * 4;
  __bf16* qb = (__bf16*)w;                 w += (size_t)HH * TT * HD * 2;
  __bf16* ktb = (__bf16*)w;                w += (size_t)HH * HD * TT * 2;
  __bf16* vb = (__bf16*)w;                 w += (size_t)HH * TT * HD * 2;
  __bf16* aob = (__bf16*)w;                w += (size_t)TT * DD * 2;

  // 1. convert inputs to bf16
  {
    int n = TT * DD;
    f32_to_bf16_kernel<<<(n + 255) / 256, 256, 0, stream>>>(x, xb, n);
    n = DD * D3;
    f32_to_bf16_kernel<<<(n + 255) / 256, 256, 0, stream>>>(W_qkv, wqkvb, n);
    n = DD * DD;
    f32_to_bf16_kernel<<<(n + 255) / 256, 256, 0, stream>>>(W_out, woutb, n);
  }
  // 2. qkv = x @ W_qkv   (4096 x 6144 x 2048)
  gemm_bf16_wmma<<<dim3(D3 / 128, TT / 128), 256, 0, stream>>>(
      xb, wqkvb, qkv, TT, D3, DD);
  // 3. l2-norm q,k; repack head-major (k transposed)
  qk_norm_kernel<<<(TT * HH) / 8, 256, 0, stream>>>(qkv, qb, ktb, vb);
  // 4. flash attention
  flash_attn_wmma<<<(HH * (TT / 16)) / 8, 256, 0, stream>>>(qb, ktb, vb,
                                                            attn_mask, aob);
  // 5. out = attn_out @ W_out  (4096 x 2048 x 2048)
  gemm_bf16_wmma<<<dim3(DD / 128, TT / 128), 256, 0, stream>>>(
      aob, woutb, outp, TT, DD, DD);

  (void)in_sizes; (void)n_in; (void)out_size; (void)ws_size;
}
